// AllenLongFormerSelfAttentionBlock_73916387164215
// MI455X (gfx1250) — compile-verified, matrix operand order checked by reading, not run on hardware
//
#include <hip/hip_runtime.h>
#include <hip/hip_bf16.h>
#include <math.h>

// ---------------------------------------------------------------------------
// Problem constants (from reference): B=2, S0=2048, D=768, H=12, d=64, w=256
// padded S = 2560, window-padded key space SP = S + 2w = 3072, chunks used = 8
// ---------------------------------------------------------------------------
#define BB     2
#define S0SZ   2048
#define SV     2560          // padded sequence
#define WSZ    256           // window
#define SP     3072          // SV + 2*WSZ (zero-padded key space)
#define DM     768           // model dim
#define HN     12            // heads
#define DH     64            // head dim
#define CUSED  8             // live chunks (rows >= S0 are dropped by reference)
#define NEGV   (-10000.0f)

typedef __attribute__((ext_vector_type(16))) __bf16 v16bf;
typedef __attribute__((ext_vector_type(8)))  __bf16 v8bf;
typedef __attribute__((ext_vector_type(8)))  float  v8f;

static __device__ inline v8f zero8() {
  v8f z;
#pragma unroll
  for (int i = 0; i < 8; ++i) z[i] = 0.0f;
  return z;
}

// Load one 16x32 bf16 WMMA operand (A-layout; identical index pattern serves
// B-layout when the source matrix is stored N-major / reduction-contiguous).
// Per ISA 7.12.2: lanes 0-15 hold K 0..7 & 16..23, lanes 16-31 hold 8..15 &
// 24..31 -> two aligned 16B (global_load_b128) segments per lane.
static __device__ inline v16bf load_tile16x32(const __bf16* __restrict__ base,
                                              int stride, int row, int kbase,
                                              int lane) {
  int r   = row + (lane & 15);
  int hi8 = (lane >> 4) << 3;                       // 0 or 8
  const __bf16* p = base + (size_t)r * stride + kbase + hi8;
  v8bf a = *(const v8bf*)p;
  v8bf b = *(const v8bf*)(p + 16);
  v16bf o;
#pragma unroll
  for (int i = 0; i < 8; ++i) { o[i] = a[i]; o[i + 8] = b[i]; }
  return o;
}

// ---------------------------------------------------------------------------
// Small utility kernels
// ---------------------------------------------------------------------------
__global__ void zero_bf16_kernel(__bf16* p, int n) {
  int i = blockIdx.x * 256 + threadIdx.x;
  if (i < n) p[i] = (__bf16)0.0f;
}

// x (B,S0,D) f32 -> xp_bf16 (B,SV,D), zero rows >= S0
__global__ void pad_cvt_x_kernel(const float* __restrict__ x,
                                 __bf16* __restrict__ xpb) {
  size_t i = (size_t)blockIdx.x * 256 + threadIdx.x;
  const size_t total = (size_t)BB * SV * DM;
  if (i >= total) return;
  int dcol = (int)(i % DM);
  size_t r = i / DM;
  int s = (int)(r % SV);
  int b = (int)(r / SV);
  float v = (s < S0SZ) ? x[((size_t)b * S0SZ + s) * DM + dcol] : 0.0f;
  xpb[i] = (__bf16)v;
}

// Wt[n*K + k] = (bf16) W[k*N + n]
__global__ void transpose_cvt_kernel(const float* __restrict__ W,
                                     __bf16* __restrict__ Wt, int K, int N) {
  size_t i = (size_t)blockIdx.x * 256 + threadIdx.x;
  if (i >= (size_t)K * N) return;
  int n = (int)(i % N);
  int k = (int)(i / N);
  Wt[(size_t)n * K + k] = (__bf16)W[i];
}

// ---------------------------------------------------------------------------
// Generic WMMA GEMM:  out = epilogue( (A_bf16 @ Wt_bf16^T + bias) * alpha )
//   A: [M][K] bf16, Bt: [N][K] bf16 (pre-transposed weights)
//   Workgroup = 256 threads = 8 waves; wave tile 16x64, block tile 128x64.
//   The shared 64x32 B slab per K-step is staged into a double-buffered LDS
//   region with global_load_async_to_lds_b128 (ASYNCcnt), overlapping the
//   global fetch with the WMMA stream; waves then read B operands from LDS.
//   bfmode: 0 plain [r*N+c]; 1 window-padded K buffer; 2 transposed vT.
// ---------------------------------------------------------------------------
__global__ __launch_bounds__(256) void gemm_wmma_kernel(
    const __bf16* __restrict__ A, const __bf16* __restrict__ Bt,
    const float* __restrict__ bias, float* __restrict__ Cf,
    __bf16* __restrict__ Cb, const float* __restrict__ residual,
    int M, int N, int K, float alpha, int do_gelu,
    int bfmode, int rows_per_batch, int pad, int padded_rows) {
  __shared__ __attribute__((aligned(16))) __bf16 bs[2][64 * 32];

  int lane = threadIdx.x & 31;
  int wave = threadIdx.x >> 5;
  int m0 = blockIdx.y * 128 + wave * 16;
  int n0 = blockIdx.x * 64;

  int tid  = threadIdx.x;
  int brow = tid >> 2;             // 0..63   (row of the 64x32 B slab)
  int bseg = (tid & 3) << 3;       // 0,8,16,24 (8-elem = 16B segment)

  // async-stage the 64x32 B slab for K-offset k into LDS buffer `buf`
  auto stage_b = [&](int buf, int k) {
    unsigned dst = (unsigned)(size_t)&bs[buf][brow * 32 + bseg];
    const __bf16* src = Bt + (size_t)(n0 + brow) * K + k + bseg;
    asm volatile("global_load_async_to_lds_b128 %0, %1, off"
                 :: "v"(dst), "v"(src) : "memory");
  };

  v8f acc[4];
#pragma unroll
  for (int t = 0; t < 4; ++t) acc[t] = zero8();

  stage_b(0, 0);

  for (int k = 0; k < K; k += 32) {
    int cur = (k >> 5) & 1;
    asm volatile("s_wait_asynccnt 0x0" ::: "memory");
    __syncthreads();                       // bs[cur] fully resident
    if (k + 32 < K) {                      // overlap next slab with compute
      stage_b(cur ^ 1, k + 32);
      __builtin_prefetch(A + (size_t)(m0 + (lane & 15)) * K + k + 32, 0, 1);
    }

    v16bf a = load_tile16x32(A, K, m0, k, lane);
    v16bf b[4];
    {
      int hi8 = (lane >> 4) << 3;
#pragma unroll
      for (int t = 0; t < 4; ++t) {
        int r = 16 * t + (lane & 15);
        v8bf lo = *(const v8bf*)&bs[cur][r * 32 + hi8];
        v8bf hi = *(const v8bf*)&bs[cur][r * 32 + 16 + hi8];
#pragma unroll
        for (int i = 0; i < 8; ++i) { b[t][i] = lo[i]; b[t][i + 8] = hi[i]; }
      }
    }
#pragma unroll
    for (int t = 0; t < 4; ++t)
      acc[t] = __builtin_amdgcn_wmma_f32_16x16x32_bf16(
          false, a, false, b[t], (short)0, acc[t], false, false);
    __syncthreads();                       // reads done before next overwrite
  }

  int col = lane & 15;
  int rowOff = (lane >> 4) * 8;  // C layout: VGPR e = row e (lo) / e+8 (hi)
#pragma unroll
  for (int t = 0; t < 4; ++t) {
#pragma unroll
    for (int e = 0; e < 8; ++e) {
      int r = m0 + e + rowOff;
      int cc = n0 + 16 * t + col;
      float v = (acc[t][e] + bias[cc]) * alpha;
      if (do_gelu) v = 0.5f * v * (1.0f + erff(v * 0.70710678118654752f));
      if (residual) v += residual[(size_t)r * N + cc];
      if (Cf) Cf[(size_t)r * N + cc] = v;
      if (Cb) {
        size_t idx;
        if (bfmode == 0) {
          idx = (size_t)r * N + cc;
        } else {
          int rb = r / rows_per_batch;
          int s = r % rows_per_batch;
          if (bfmode == 1)  // K: [b][pad + s][N]
            idx = ((size_t)rb * padded_rows + pad + s) * N + cc;
          else              // V^T: [b][cc][pad + s]
            idx = ((size_t)rb * N + cc) * padded_rows + pad + s;
        }
        Cb[idx] = (__bf16)v;
      }
    }
  }
}

// ---------------------------------------------------------------------------
// Windowed attention.  Workgroup = 128 threads (4 waves); each wave handles a
// 16-row query tile of chunk c, head h, batch b.  Two-pass softmax:
//   pass 1: WMMA scores -> masked row max (16-lane shfl_xor reduction)
//   pass 2: WMMA scores -> p = exp(s-m); P staged via wave-private LDS to
//           A-layout bf16; P@V via WMMA against vT (window-zero-padded).
// Global key column (k[:,g], v[:,g]) folded in scalarly.
// ---------------------------------------------------------------------------
__global__ __launch_bounds__(128) void attn_win_kernel(
    const __bf16* __restrict__ qbf,   // [B][SV][DM]   (scale folded in)
    const __bf16* __restrict__ kpad,  // [B][SP][DM]   (±w zero rows)
    const __bf16* __restrict__ vT,    // [B][DM][SP]   (±w zero cols)
    float* __restrict__ out) {        // [B][S0][DM]
  int c = blockIdx.x >> 2;       // chunk 0..7 (live rows only)
  int rb = blockIdx.x & 3;       // 64-row block within chunk
  int h = blockIdx.y;
  int b = blockIdx.z;
  int wave = threadIdx.x >> 5;
  int lane = threadIdx.x & 31;
  int x0 = rb * 64 + wave * 16;  // query row within chunk [0,256)
  int col = lane & 15;
  int rowOff = (lane >> 4) * 8;

  const __bf16* qb = qbf + (size_t)b * SV * DM + h * DH;
  const __bf16* kb = kpad + (size_t)b * SP * DM + h * DH;
  const __bf16* vb = vT + ((size_t)b * DM + h * DH) * SP;

  // Q operand (16 rows x K=64) once: two 16x32 A tiles
  v16bf qa0 = load_tile16x32(qb, DM, c * WSZ + x0, 0, lane);
  v16bf qa1 = load_tile16x32(qb, DM, c * WSZ + x0, 32, lane);

  // global-column scores: sg[e] = q_row . k[g]  (g padded index = SP-257)
  const int GPAD = WSZ + SV - 1;  // 2815
  float sg[8];
#pragma unroll
  for (int e = 0; e < 8; ++e) {
    int r = c * WSZ + x0 + e + rowOff;
    float a = 0.0f;
    for (int f = 0; f < DH; ++f)
      a += (float)qb[(size_t)r * DM + f] * (float)kb[(size_t)GPAD * DM + f];
    sg[e] = a;
  }

  // -------- pass 1: masked row max --------
  float mrow[8];
#pragma unroll
  for (int e = 0; e < 8; ++e) mrow[e] = sg[e];

  for (int yc = 0; yc < 3 * WSZ; yc += 32) {
    v16bf kt[4];
#pragma unroll
    for (int half = 0; half < 2; ++half) {
      kt[2 * half + 0] = load_tile16x32(kb, DM, c * WSZ + yc + 16 * half, 0, lane);
      kt[2 * half + 1] = load_tile16x32(kb, DM, c * WSZ + yc + 16 * half, 32, lane);
    }
    v8f s[2];
    s[0] = zero8(); s[1] = zero8();
#pragma unroll
    for (int half = 0; half < 2; ++half) {
      s[half] = __builtin_amdgcn_wmma_f32_16x16x32_bf16(
          false, qa0, false, kt[2 * half + 0], (short)0, s[half], false, false);
      s[half] = __builtin_amdgcn_wmma_f32_16x16x32_bf16(
          false, qa1, false, kt[2 * half + 1], (short)0, s[half], false, false);
    }
#pragma unroll
    for (int half = 0; half < 2; ++half) {
      int y = yc + 16 * half + col;
      int j = c * WSZ - WSZ + y;
#pragma unroll
      for (int e = 0; e < 8; ++e) {
        int xl = x0 + e + rowOff;
        int rel = y - xl;
        bool ok = (rel >= 0) & (rel <= 2 * WSZ) & (j >= 0) & (j < SV);
        float sv = s[half][e] + ((j == SV - 1) ? NEGV : 0.0f);
        if (ok) mrow[e] = fmaxf(mrow[e], sv);
      }
    }
  }
#pragma unroll
  for (int e = 0; e < 8; ++e)
    for (int m = 1; m < 16; m <<= 1)
      mrow[e] = fmaxf(mrow[e], __shfl_xor(mrow[e], m, 32));

  // -------- pass 2: exp / sum / P@V --------
  __shared__ __attribute__((aligned(16))) __bf16 pbuf[4][16 * 32];
  __bf16* pw = &pbuf[wave][0];
  float psum[8];
#pragma unroll
  for (int e = 0; e < 8; ++e) psum[e] = 0.0f;
  v8f acc[4];
#pragma unroll
  for (int t = 0; t < 4; ++t) acc[t] = zero8();

  for (int yc = 0; yc < 3 * WSZ; yc += 32) {
    v16bf kt[4];
#pragma unroll
    for (int half = 0; half < 2; ++half) {
      kt[2 * half + 0] = load_tile16x32(kb, DM, c * WSZ + yc + 16 * half, 0, lane);
      kt[2 * half + 1] = load_tile16x32(kb, DM, c * WSZ + yc + 16 * half, 32, lane);
    }
    v8f s[2];
    s[0] = zero8(); s[1] = zero8();
#pragma unroll
    for (int half = 0; half < 2; ++half) {
      s[half] = __builtin_amdgcn_wmma_f32_16x16x32_bf16(
          false, qa0, false, kt[2 * half + 0], (short)0, s[half], false, false);
      s[half] = __builtin_amdgcn_wmma_f32_16x16x32_bf16(
          false, qa1, false, kt[2 * half + 1], (short)0, s[half], false, false);
    }
    // masked exp, stage P tile (C-layout -> LDS row-major bf16)
#pragma unroll
    for (int half = 0; half < 2; ++half) {
      int y = yc + 16 * half + col;
      int j = c * WSZ - WSZ + y;
#pragma unroll
      for (int e = 0; e < 8; ++e) {
        int xl = x0 + e + rowOff;
        int rel = y - xl;
        bool ok = (rel >= 0) & (rel <= 2 * WSZ) & (j >= 0) & (j < SV);
        float sv = s[half][e] + ((j == SV - 1) ? NEGV : 0.0f);
        float p = ok ? __expf(sv - mrow[e]) : 0.0f;
        psum[e] += p;
        pw[(e + rowOff) * 32 + 16 * half + col] = (__bf16)p;  // ds_store_b16
      }
    }
    asm volatile("s_wait_dscnt 0x0" ::: "memory");  // wave-local LDS ordering
    // re-read P in WMMA A-layout (two ds_load_b128 per lane)
    v16bf pa;
    {
      int m = lane & 15;
      int hi8 = (lane >> 4) << 3;
      v8bf a = *(const v8bf*)&pw[m * 32 + hi8];
      v8bf bq = *(const v8bf*)&pw[m * 32 + 16 + hi8];
#pragma unroll
      for (int i = 0; i < 8; ++i) { pa[i] = a[i]; pa[i + 8] = bq[i]; }
    }
    v16bf vop[4];
#pragma unroll
    for (int t = 0; t < 4; ++t)
      vop[t] = load_tile16x32(vb, SP, 16 * t, c * WSZ + yc, lane);
#pragma unroll
    for (int t = 0; t < 4; ++t)
      acc[t] = __builtin_amdgcn_wmma_f32_16x16x32_bf16(
          false, pa, false, vop[t], (short)0, acc[t], false, false);
  }
#pragma unroll
  for (int e = 0; e < 8; ++e)
    for (int m = 1; m < 16; m <<= 1) psum[e] += __shfl_xor(psum[e], m, 32);

  // finalize: add global column, normalize, scatter-store
#pragma unroll
  for (int t = 0; t < 4; ++t) {
    int f = 16 * t + col;
    float vg = (float)vb[(size_t)f * SP + GPAD];
#pragma unroll
    for (int e = 0; e < 8; ++e) {
      int row = c * WSZ + x0 + e + rowOff;  // < 2048 by construction
      float pg = __expf(sg[e] - mrow[e]);
      float val = (acc[t][e] + pg * vg) / (psum[e] + pg);
      out[((size_t)b * S0SZ + row) * DM + h * DH + f] = val;
    }
  }
}

// ---------------------------------------------------------------------------
// Residual + LayerNorm:  xn = LN(x + attn) * gamma + beta   (rows = B*S0)
// ---------------------------------------------------------------------------
__global__ __launch_bounds__(256) void ln_kernel(
    const float* __restrict__ x, const float* __restrict__ attn,
    const float* __restrict__ gam, const float* __restrict__ bet,
    float* __restrict__ xn, __bf16* __restrict__ xnb) {
  int row = blockIdx.x;
  __shared__ float red[256];
  float vals[3];
  float s = 0.0f;
  const float* xr = x + (size_t)row * DM;
  const float* ar = attn + (size_t)row * DM;
#pragma unroll
  for (int i = 0; i < 3; ++i) {
    vals[i] = xr[threadIdx.x + 256 * i] + ar[threadIdx.x + 256 * i];
    s += vals[i];
  }
  red[threadIdx.x] = s;
  __syncthreads();
  for (int off = 128; off > 0; off >>= 1) {
    if (threadIdx.x < off) red[threadIdx.x] += red[threadIdx.x + off];
    __syncthreads();
  }
  float mu = red[0] / (float)DM;
  __syncthreads();
  float s2 = 0.0f;
#pragma unroll
  for (int i = 0; i < 3; ++i) { float d = vals[i] - mu; s2 += d * d; }
  red[threadIdx.x] = s2;
  __syncthreads();
  for (int off = 128; off > 0; off >>= 1) {
    if (threadIdx.x < off) red[threadIdx.x] += red[threadIdx.x + off];
    __syncthreads();
  }
  float rstd = rsqrtf(red[0] / (float)DM + 1e-5f);
#pragma unroll
  for (int i = 0; i < 3; ++i) {
    int cc = threadIdx.x + 256 * i;
    float v = (vals[i] - mu) * rstd * gam[cc] + bet[cc];
    xn[(size_t)row * DM + cc] = v;
    xnb[(size_t)row * DM + cc] = (__bf16)v;
  }
}

// ---------------------------------------------------------------------------
// Host-side orchestration
// ---------------------------------------------------------------------------
extern "C" void kernel_launch(void* const* d_in, const int* in_sizes, int n_in,
                              void* d_out, int out_size, void* d_ws,
                              size_t ws_size, hipStream_t stream) {
  (void)in_sizes; (void)n_in; (void)out_size; (void)ws_size;
  const float* x  = (const float*)d_in[0];
  const float* Wq = (const float*)d_in[2];  const float* bq = (const float*)d_in[3];
  const float* Wk = (const float*)d_in[4];  const float* bk = (const float*)d_in[5];
  const float* Wv = (const float*)d_in[6];  const float* bv = (const float*)d_in[7];
  // d_in[8..13] (Wqg/bqg/Wkg/bkg/Wvg/bvg) only affect output row g >= S0 -> dead
  const float* lng = (const float*)d_in[14]; const float* lnb = (const float*)d_in[15];
  const float* W1 = (const float*)d_in[16]; const float* b1 = (const float*)d_in[17];
  const float* W2 = (const float*)d_in[18]; const float* b2 = (const float*)d_in[19];
  float* outp = (float*)d_out;

  // workspace carve (256B aligned)
  char* wsp = (char*)d_ws;
  auto alloc = [&](size_t bytes) {
    void* p = (void*)wsp;
    wsp += (bytes + 255) & ~(size_t)255;
    return p;
  };
  __bf16* xpb  = (__bf16*)alloc((size_t)BB * SV * DM * 2);   // padded x, bf16
  __bf16* wqT  = (__bf16*)alloc((size_t)DM * DM * 2);
  __bf16* wkT  = (__bf16*)alloc((size_t)DM * DM * 2);
  __bf16* wvT  = (__bf16*)alloc((size_t)DM * DM * 2);
  __bf16* w1T  = (__bf16*)alloc((size_t)4 * DM * DM * 2);    // [3072][768]
  __bf16* w2T  = (__bf16*)alloc((size_t)4 * DM * DM * 2);    // [768][3072]
  __bf16* qb   = (__bf16*)alloc((size_t)BB * SV * DM * 2);
  __bf16* kpad = (__bf16*)alloc((size_t)BB * SP * DM * 2);
  __bf16* vTp  = (__bf16*)alloc((size_t)BB * DM * SP * 2);
  float*  attn = (float*)alloc((size_t)BB * S0SZ * DM * 4);
  float*  xn   = (float*)alloc((size_t)BB * S0SZ * DM * 4);
  __bf16* xnb  = (__bf16*)alloc((size_t)BB * S0SZ * DM * 2);
  __bf16* hb   = (__bf16*)alloc((size_t)BB * S0SZ * (4 * DM) * 2);

  // 1) zero window pads (GEMMs fill the interiors)
  {
    int nk = BB * SP * DM;
    zero_bf16_kernel<<<(nk + 255) / 256, 256, 0, stream>>>(kpad, nk);
    zero_bf16_kernel<<<(nk + 255) / 256, 256, 0, stream>>>(vTp, nk);
  }
  // 2) pad + convert activations, transpose + convert live weights
  {
    size_t n = (size_t)BB * SV * DM;
    pad_cvt_x_kernel<<<(unsigned)((n + 255) / 256), 256, 0, stream>>>(x, xpb);
  }
  transpose_cvt_kernel<<<(DM * DM + 255) / 256, 256, 0, stream>>>(Wq, wqT, DM, DM);
  transpose_cvt_kernel<<<(DM * DM + 255) / 256, 256, 0, stream>>>(Wk, wkT, DM, DM);
  transpose_cvt_kernel<<<(DM * DM + 255) / 256, 256, 0, stream>>>(Wv, wvT, DM, DM);
  transpose_cvt_kernel<<<(4 * DM * DM + 255) / 256, 256, 0, stream>>>(W1, w1T, DM, 4 * DM);
  transpose_cvt_kernel<<<(4 * DM * DM + 255) / 256, 256, 0, stream>>>(W2, w2T, 4 * DM, DM);

  const int Mqkv = BB * SV;  // 5120
  // 3) Q = (x@Wq + bq) * 1/sqrt(64)       -> qb   (plain bf16)
  gemm_wmma_kernel<<<dim3(DM / 64, Mqkv / 128), 256, 0, stream>>>(
      xpb, wqT, bq, nullptr, qb, nullptr, Mqkv, DM, DM, 0.125f, 0, 0, 0, 0, 0);
  // 4) K = x@Wk + bk                      -> kpad (window-padded bf16)
  gemm_wmma_kernel<<<dim3(DM / 64, Mqkv / 128), 256, 0, stream>>>(
      xpb, wkT, bk, nullptr, kpad, nullptr, Mqkv, DM, DM, 1.0f, 0, 1, SV, WSZ, SP);
  // 5) V = x@Wv + bv                      -> vTp  (transposed window-padded)
  gemm_wmma_kernel<<<dim3(DM / 64, Mqkv / 128), 256, 0, stream>>>(
      xpb, wvT, bv, nullptr, vTp, nullptr, Mqkv, DM, DM, 1.0f, 0, 2, SV, WSZ, SP);
  // 6) windowed attention (live chunks only)
  attn_win_kernel<<<dim3(CUSED * 4, HN, BB), 128, 0, stream>>>(qb, kpad, vTp, attn);
  // 7) residual + layernorm
  ln_kernel<<<BB * S0SZ, 256, 0, stream>>>(x, attn, lng, lnb, xn, xnb);
  // 8) FFN: h = gelu(xn@W1 + b1)
  const int Mff = BB * S0SZ;  // 4096
  gemm_wmma_kernel<<<dim3((4 * DM) / 64, Mff / 128), 256, 0, stream>>>(
      xnb, w1T, b1, nullptr, hb, nullptr, Mff, 4 * DM, DM, 1.0f, 1, 0, 0, 0, 0);
  // 9) out = xn + h@W2 + b2
  gemm_wmma_kernel<<<dim3(DM / 64, Mff / 128), 256, 0, stream>>>(
      hb, w2T, b2, outp, nullptr, xn, Mff, DM, 4 * DM, 1.0f, 0, 0, 0, 0, 0);
}
